// ModalAttn_65893388255513
// MI455X (gfx1250) — matrix operand order repaired and checked by reading, not compiled, not run on hardware
//
#include <hip/hip_runtime.h>

typedef __attribute__((ext_vector_type(16))) _Float16 v16h;
typedef __attribute__((ext_vector_type(8)))  _Float16 v8h;
typedef __attribute__((ext_vector_type(8)))  float    v8f;

#define E_DIM  512
#define HEADS  8
#define HD_DIM 64
#define QKV3   1536

__device__ __forceinline__ v8f wmma_f16(v16h a, v16h b, v8f c) {
  return __builtin_amdgcn_wmma_f32_16x16x32_f16(
      /*neg_a=*/false, a, /*neg_b=*/false, b,
      /*c_mod=*/(short)0, c, /*reuse_a=*/false, /*reuse_b=*/false);
}

__device__ __forceinline__ v16h cat8(v8h lo, v8h hi) {
  return __builtin_shufflevector(lo, hi, 0, 1, 2, 3, 4, 5, 6, 7,
                                 8, 9, 10, 11, 12, 13, 14, 15);
}

__device__ __forceinline__ v8h cvt8(float4 a, float4 b) {
  v8h r;
  r[0] = (_Float16)a.x; r[1] = (_Float16)a.y; r[2] = (_Float16)a.z; r[3] = (_Float16)a.w;
  r[4] = (_Float16)b.x; r[5] = (_Float16)b.y; r[6] = (_Float16)b.z; r[7] = (_Float16)b.w;
  return r;
}

// fragment fetch + 4 WMMAs for one 32-deep K step (shared by main/remainder)
__device__ __forceinline__ void frag_mma(
    const _Float16 (*As)[40], const _Float16 (*Bs)[40],
    int wm, int wn, int lg, int ln,
    v8f& acc00, v8f& acc01, v8f& acc10, v8f& acc11)
{
  const _Float16* ar0 = &As[wm * 32 + ln][0];
  const _Float16* ar1 = &As[wm * 32 + 16 + ln][0];
  v16h a0 = cat8(*(const v8h*)(ar0 + lg * 8), *(const v8h*)(ar0 + 16 + lg * 8));
  v16h a1 = cat8(*(const v8h*)(ar1 + lg * 8), *(const v8h*)(ar1 + 16 + lg * 8));
  const _Float16* br0 = &Bs[wn * 32 + ln][0];
  const _Float16* br1 = &Bs[wn * 32 + 16 + ln][0];
  v16h b0 = cat8(*(const v8h*)(br0 + lg * 8), *(const v8h*)(br0 + 16 + lg * 8));
  v16h b1 = cat8(*(const v8h*)(br1 + lg * 8), *(const v8h*)(br1 + 16 + lg * 8));
  acc00 = wmma_f16(a0, b0, acc00);
  acc01 = wmma_f16(a0, b1, acc01);
  acc10 = wmma_f16(a1, b0, acc10);
  acc11 = wmma_f16(a1, b1, acc11);
}

// ---------------------------------------------------------------------------
// Generic GEMM: C[M,Nmat] = act( (A[M,K] @ B[K,Nmat] + bias) * alpha )
// f32 in memory -> f16 in LDS -> WMMA f32 accumulate.
// Block: 128 threads (4 waves), 64x64 tile, each wave a 32x32 quadrant.
// Guard-free main K loop (pointer increments) + 16-wide remainder (K=16 case).
// ---------------------------------------------------------------------------
__global__ void __launch_bounds__(128) gemm_wmma_kernel(
    const float* __restrict__ A, int lda,
    const float* __restrict__ Bm, int ldb,
    const float* __restrict__ bias,
    void* __restrict__ Cout, int ldc,
    int M, int K, int Nmat, float alpha, int relu, int half_out)
{
  __shared__ _Float16 As[64][40];   // [m][k]
  __shared__ _Float16 Bs[64][40];   // [n][k] transposed

  const int tid  = threadIdx.x;
  const int w    = tid >> 5;
  const int lane = tid & 31;
  const int lg   = lane >> 4;
  const int ln   = lane & 15;
  const int wm   = w >> 1;
  const int wn   = w & 1;
  const int bm   = blockIdx.y * 64;
  const int bn   = blockIdx.x * 64;

  const int am  = tid >> 1, akc = (tid & 1) * 16;
  const int bnc = tid & 63, bkc = (tid >> 6) * 16;
  const int gmA = (bm + am < M) ? bm + am : M - 1;        // clamp: masked later
  const int gnB = (bn + bnc < Nmat) ? bn + bnc : Nmat - 1;

  const float* aPtr = A + (size_t)gmA * lda + akc;        // += 32 per step
  const float* bPtr = Bm + (size_t)bkc * ldb + gnB;       // += 32*ldb per step
  const size_t bStep = (size_t)32 * ldb;

  v8f acc00 = {}, acc01 = {}, acc10 = {}, acc11 = {};

  const int kfull = K & ~31;
  for (int k0 = 0; k0 < kfull; k0 += 32) {
    // prefetch next tiles (speculative; dropped if past end)
    __builtin_prefetch(aPtr + 32, 0, 1);
    __builtin_prefetch(bPtr + bStep, 0, 1);
    // ---- stage A (4x b128 loads) ----
    {
      const float4* p = (const float4*)aPtr;
      *(v8h*)&As[am][akc]     = cvt8(p[0], p[1]);
      *(v8h*)&As[am][akc + 8] = cvt8(p[2], p[3]);
    }
    // ---- stage B transposed (coalesced strided loads) ----
    {
      v8h h0, h1;
#pragma unroll
      for (int j = 0; j < 8; ++j) h0[j] = (_Float16)bPtr[(size_t)j * ldb];
#pragma unroll
      for (int j = 0; j < 8; ++j) h1[j] = (_Float16)bPtr[(size_t)(8 + j) * ldb];
      *(v8h*)&Bs[bnc][bkc]     = h0;
      *(v8h*)&Bs[bnc][bkc + 8] = h1;
    }
    aPtr += 32;
    bPtr += bStep;
    __syncthreads();
    frag_mma(As, Bs, wm, wn, lg, ln, acc00, acc01, acc10, acc11);
    __syncthreads();
  }

  if (K & 31) {   // 16-wide remainder (only the K=16 GEMM takes this)
    {
      v8h h0 = {}, h1 = {};
      if (akc == 0) {
        const float4* p = (const float4*)aPtr;
        h0 = cvt8(p[0], p[1]);
        h1 = cvt8(p[2], p[3]);
      }
      *(v8h*)&As[am][akc]     = h0;
      *(v8h*)&As[am][akc + 8] = h1;
    }
    {
      v8h h0 = {}, h1 = {};
      if (bkc == 0) {
#pragma unroll
        for (int j = 0; j < 8; ++j) h0[j] = (_Float16)bPtr[(size_t)j * ldb];
#pragma unroll
        for (int j = 0; j < 8; ++j) h1[j] = (_Float16)bPtr[(size_t)(8 + j) * ldb];
      }
      *(v8h*)&Bs[bnc][bkc]     = h0;
      *(v8h*)&Bs[bnc][bkc + 8] = h1;
    }
    __syncthreads();
    frag_mma(As, Bs, wm, wn, lg, ln, acc00, acc01, acc10, acc11);
    __syncthreads();
  }

  // ---- epilogue ----
  const int c0 = bn + wn * 32 + ln;
  const int c1 = c0 + 16;
  const float bv0 = (c0 < Nmat) ? bias[c0] : 0.f;
  const float bv1 = (c1 < Nmat) ? bias[c1] : 0.f;
#pragma unroll
  for (int t = 0; t < 2; ++t) {
    v8f va = t ? acc10 : acc00;
    v8f vb = t ? acc11 : acc01;
#pragma unroll
    for (int r = 0; r < 8; ++r) {
      int row = bm + wm * 32 + t * 16 + lg * 8 + r;
      if (row >= M) continue;
      float x0 = (va[r] + bv0) * alpha;
      float x1 = (vb[r] + bv1) * alpha;
      if (relu) { x0 = fmaxf(x0, 0.f); x1 = fmaxf(x1, 0.f); }
      if (half_out) {
        _Float16* C = (_Float16*)Cout;
        if (c0 < Nmat) C[(size_t)row * ldc + c0] = (_Float16)x0;
        if (c1 < Nmat) C[(size_t)row * ldc + c1] = (_Float16)x1;
      } else {
        float* C = (float*)Cout;
        if (c0 < Nmat) C[(size_t)row * ldc + c0] = x0;
        if (c1 < Nmat) C[(size_t)row * ldc + c1] = x1;
      }
    }
  }
}

// ---------------------------------------------------------------------------
__global__ void concat_img_kernel(const float* __restrict__ img,
                                  float* __restrict__ out, int N)
{
  int i = blockIdx.x * blockDim.x + threadIdx.x;
  if (i < N * 64) out[(size_t)(i >> 6) * 128 + 64 + (i & 63)] = img[i];
}

// ---------------------------------------------------------------------------
// Flash attention over ragged batches.
// qkvp: [N,1536] f16 (q|k|v), q pre-scaled by 1/sqrt(HD).
// Block: 128 thr (4 waves) = 64 query rows for one (batch, head).
// Clamp/mask-free main loop over full 32-key tiles; one tail tile with
// clamped addresses + column masking (S=-1e30 -> p=0 annihilates garbage).
// ---------------------------------------------------------------------------
__global__ void __launch_bounds__(128) flash_attn_kernel(
    const _Float16* __restrict__ qkvp, const int* __restrict__ lens,
    float* __restrict__ ctx)
{
  __shared__ _Float16 Ks[32][72];       // [key][d]
  __shared__ _Float16 Vs[64][40];       // [d][key] transposed
  __shared__ _Float16 Ps[4][16][40];    // per-wave P tile [qrow][key]

  const int b = blockIdx.z;
  const int h = blockIdx.y;
  int offb = 0;
  for (int i = 0; i < b; ++i) offb += lens[i];
  const int len = lens[b];
  const int q0  = blockIdx.x * 64;
  if (q0 >= len) return;

  const int tid  = threadIdx.x;
  const int w    = tid >> 5;
  const int lane = tid & 31;
  const int lg   = lane >> 4;
  const int ln   = lane & 15;

  // Q fragments: clamp row, contiguous v8h loads
  v16h qa0, qa1;
  {
    int lq = q0 + w * 16 + ln;
    if (lq >= len) lq = len - 1;
    const _Float16* qp = qkvp + (size_t)(offb + lq) * QKV3 + h * HD_DIM;
    qa0 = cat8(*(const v8h*)(qp + lg * 8), *(const v8h*)(qp + 16 + lg * 8));
    qa1 = cat8(*(const v8h*)(qp + 32 + lg * 8), *(const v8h*)(qp + 48 + lg * 8));
  }

  float mrow[8], lrow[8];
  v8f acc0 = {}, acc1 = {}, acc2 = {}, acc3 = {};
#pragma unroll
  for (int r = 0; r < 8; ++r) { mrow[r] = -3.0e38f; lrow[r] = 0.f; }

  // staging coords
  const int skey = tid >> 2, skc = (tid & 3) * 16;   // K: key row, d chunk
  const int svd  = tid & 63, svk = (tid >> 6) * 16;  // V: d column, key chunk

  // linear staging pointers for the main (full-tile) loop
  const _Float16* kPtr = qkvp + ((size_t)(offb + skey)) * QKV3 + E_DIM + h * HD_DIM + skc;
  const _Float16* vPtr = qkvp + ((size_t)(offb + svk)) * QKV3 + 2 * E_DIM + h * HD_DIM + svd;
  const size_t tStep = (size_t)32 * QKV3;

  const int ktfull = len & ~31;

  for (int kt = 0; kt < ktfull; kt += 32) {
    __builtin_prefetch(kPtr + tStep, 0, 1);
    __builtin_prefetch(vPtr + tStep, 0, 1);
    // ---- stage K [key][d] ----
    *(v8h*)&Ks[skey][skc]     = *(const v8h*)kPtr;
    *(v8h*)&Ks[skey][skc + 8] = *(const v8h*)(kPtr + 8);
    // ---- stage V transposed [d][key] (immediate-offset strided loads) ----
    {
      v8h h0, h1;
#pragma unroll
      for (int j = 0; j < 8; ++j) h0[j] = vPtr[(size_t)j * QKV3];
#pragma unroll
      for (int j = 0; j < 8; ++j) h1[j] = vPtr[(size_t)(8 + j) * QKV3];
      *(v8h*)&Vs[svd][svk]     = h0;
      *(v8h*)&Vs[svd][svk + 8] = h1;
    }
    kPtr += tStep;
    vPtr += tStep;
    __syncthreads();

    // ---- S = Q @ K^T (no masking: all 32 keys valid) ----
    v8f Sa = {}, Sb = {};
    {
      const _Float16* kr0 = &Ks[ln][0];
      const _Float16* kr1 = &Ks[16 + ln][0];
      v16h f0 = cat8(*(const v8h*)(kr0 + lg * 8), *(const v8h*)(kr0 + 16 + lg * 8));
      v16h f1 = cat8(*(const v8h*)(kr0 + 32 + lg * 8), *(const v8h*)(kr0 + 48 + lg * 8));
      Sa = wmma_f16(qa0, f0, Sa);
      Sa = wmma_f16(qa1, f1, Sa);
      v16h g0 = cat8(*(const v8h*)(kr1 + lg * 8), *(const v8h*)(kr1 + 16 + lg * 8));
      v16h g1 = cat8(*(const v8h*)(kr1 + 32 + lg * 8), *(const v8h*)(kr1 + 48 + lg * 8));
      Sb = wmma_f16(qa0, g0, Sb);
      Sb = wmma_f16(qa1, g1, Sb);
    }

    // ---- online softmax ----
#pragma unroll
    for (int r = 0; r < 8; ++r) {
      float t = fmaxf(Sa[r], Sb[r]);
#pragma unroll
      for (int m = 1; m < 16; m <<= 1) t = fmaxf(t, __shfl_xor(t, m, 32));
      float mn  = fmaxf(mrow[r], t);
      float scl = __expf(mrow[r] - mn);
      mrow[r] = mn;
      float pa = __expf(Sa[r] - mn);
      float pb = __expf(Sb[r] - mn);
      float rs = pa + pb;
#pragma unroll
      for (int m = 1; m < 16; m <<= 1) rs += __shfl_xor(rs, m, 32);
      lrow[r] = lrow[r] * scl + rs;
      acc0[r] *= scl; acc1[r] *= scl; acc2[r] *= scl; acc3[r] *= scl;
      Ps[w][lg * 8 + r][ln]      = (_Float16)pa;
      Ps[w][lg * 8 + r][16 + ln] = (_Float16)pb;
    }
    __syncthreads();

    // ---- O += P @ V ----
    {
      const _Float16* pr = &Ps[w][ln][0];
      v16h pf = cat8(*(const v8h*)(pr + lg * 8), *(const v8h*)(pr + 16 + lg * 8));
#pragma unroll
      for (int nb = 0; nb < 4; ++nb) {
        const _Float16* vr = &Vs[nb * 16 + ln][0];
        v16h vf = cat8(*(const v8h*)(vr + lg * 8), *(const v8h*)(vr + 16 + lg * 8));
        v8f* accp = (nb == 0) ? &acc0 : (nb == 1) ? &acc1 : (nb == 2) ? &acc2 : &acc3;
        *accp = wmma_f16(pf, vf, *accp);
      }
    }
    __syncthreads();
  }

  // ---- tail tile (len % 32 keys) ----
  if (len & 31) {
    const int kt = ktfull;
    {
      int krow = kt + skey; if (krow >= len) krow = len - 1;
      const _Float16* kp = qkvp + (size_t)(offb + krow) * QKV3 + E_DIM + h * HD_DIM + skc;
      *(v8h*)&Ks[skey][skc]     = *(const v8h*)kp;
      *(v8h*)&Ks[skey][skc + 8] = *(const v8h*)(kp + 8);
    }
    {
      v8h h0, h1;
#pragma unroll
      for (int j = 0; j < 8; ++j) {
        int r0 = kt + svk + j;     if (r0 >= len) r0 = len - 1;
        int r1 = kt + svk + 8 + j; if (r1 >= len) r1 = len - 1;
        h0[j] = qkvp[(size_t)(offb + r0) * QKV3 + 2 * E_DIM + h * HD_DIM + svd];
        h1[j] = qkvp[(size_t)(offb + r1) * QKV3 + 2 * E_DIM + h * HD_DIM + svd];
      }
      *(v8h*)&Vs[svd][svk]     = h0;
      *(v8h*)&Vs[svd][svk + 8] = h1;
    }
    __syncthreads();

    v8f Sa = {}, Sb = {};
    {
      const _Float16* kr0 = &Ks[ln][0];
      const _Float16* kr1 = &Ks[16 + ln][0];
      v16h f0 = cat8(*(const v8h*)(kr0 + lg * 8), *(const v8h*)(kr0 + 16 + lg * 8));
      v16h f1 = cat8(*(const v8h*)(kr0 + 32 + lg * 8), *(const v8h*)(kr0 + 48 + lg * 8));
      Sa = wmma_f16(qa0, f0, Sa);
      Sa = wmma_f16(qa1, f1, Sa);
      v16h g0 = cat8(*(const v8h*)(kr1 + lg * 8), *(const v8h*)(kr1 + 16 + lg * 8));
      v16h g1 = cat8(*(const v8h*)(kr1 + 32 + lg * 8), *(const v8h*)(kr1 + 48 + lg * 8));
      Sb = wmma_f16(qa0, g0, Sb);
      Sb = wmma_f16(qa1, g1, Sb);
    }
    if (kt + ln >= len) {
#pragma unroll
      for (int r = 0; r < 8; ++r) Sa[r] = -1.0e30f;
    }
    if (kt + 16 + ln >= len) {
#pragma unroll
      for (int r = 0; r < 8; ++r) Sb[r] = -1.0e30f;
    }
#pragma unroll
    for (int r = 0; r < 8; ++r) {
      float t = fmaxf(Sa[r], Sb[r]);
#pragma unroll
      for (int m = 1; m < 16; m <<= 1) t = fmaxf(t, __shfl_xor(t, m, 32));
      float mn  = fmaxf(mrow[r], t);
      float scl = __expf(mrow[r] - mn);
      mrow[r] = mn;
      float pa = __expf(Sa[r] - mn);
      float pb = __expf(Sb[r] - mn);
      float rs = pa + pb;
#pragma unroll
      for (int m = 1; m < 16; m <<= 1) rs += __shfl_xor(rs, m, 32);
      lrow[r] = lrow[r] * scl + rs;
      acc0[r] *= scl; acc1[r] *= scl; acc2[r] *= scl; acc3[r] *= scl;
      Ps[w][lg * 8 + r][ln]      = (_Float16)pa;
      Ps[w][lg * 8 + r][16 + ln] = (_Float16)pb;
    }
    __syncthreads();
    {
      const _Float16* pr = &Ps[w][ln][0];
      v16h pf = cat8(*(const v8h*)(pr + lg * 8), *(const v8h*)(pr + 16 + lg * 8));
#pragma unroll
      for (int nb = 0; nb < 4; ++nb) {
        const _Float16* vr = &Vs[nb * 16 + ln][0];
        v16h vf = cat8(*(const v8h*)(vr + lg * 8), *(const v8h*)(vr + 16 + lg * 8));
        v8f* accp = (nb == 0) ? &acc0 : (nb == 1) ? &acc1 : (nb == 2) ? &acc2 : &acc3;
        *accp = wmma_f16(pf, vf, *accp);
      }
    }
  }

  // ---- finalize ----
#pragma unroll
  for (int r = 0; r < 8; ++r) {
    int lq = q0 + w * 16 + lg * 8 + r;
    if (lq < len) {
      float inv = 1.f / lrow[r];
      size_t o = (size_t)(offb + lq) * E_DIM + h * HD_DIM;
      ctx[o + ln]      = acc0[r] * inv;
      ctx[o + 16 + ln] = acc1[r] * inv;
      ctx[o + 32 + ln] = acc2[r] * inv;
      ctx[o + 48 + ln] = acc3[r] * inv;
    }
  }
}

// ---------------------------------------------------------------------------
extern "C" void kernel_launch(void* const* d_in, const int* in_sizes, int n_in,
                              void* d_out, int out_size, void* d_ws, size_t ws_size,
                              hipStream_t stream)
{
  const float* img   = (const float*)d_in[0];
  const float* pc    = (const float*)d_in[1];
  const int*   masks = (const int*)d_in[2];
  const float* w_pc1 = (const float*)d_in[3];
  const float* b_pc1 = (const float*)d_in[4];
  const float* w_pc2 = (const float*)d_in[5];
  const float* b_pc2 = (const float*)d_in[6];
  const float* w_qkv = (const float*)d_in[7];
  const float* b_qkv = (const float*)d_in[8];
  const float* w_in  = (const float*)d_in[9];
  const float* b_in  = (const float*)d_in[10];
  const float* w_out = (const float*)d_in[11];
  const float* b_out = (const float*)d_in[12];
  const float* w_cls = (const float*)d_in[13];
  const float* b_cls = (const float*)d_in[14];

  const int N  = in_sizes[0] / 64;   // tokens
  const int Bn = in_sizes[2];        // batches

  char* ws = (char*)d_ws;
  size_t off = 0;
  auto alloc = [&](size_t bytes) {
    char* p = ws + off;
    off += (bytes + 255) & ~(size_t)255;
    return p;
  };
  float*    ws_h    = (float*)alloc((size_t)N * E_DIM * sizeof(float));
  float*    ws_qkv  = (float*)alloc((size_t)N * QKV3 * sizeof(float));
  _Float16* ws_qkvp = (_Float16*)alloc((size_t)N * QKV3 * sizeof(_Float16));
  float*    ws_ctx  = ws_h;     // reuse (h dead after pc2 GEMM)
  float*    ws_ao   = ws_qkv;   // reuse (qkv dead after projections)

  float* xm    = (float*)d_out;                    // [N,128]
  float* preds = (float*)d_out + (size_t)N * 128;  // [N,19]

  dim3 blk(128);
#define GRID64(M_, N_) dim3(((N_) + 63) / 64, ((M_) + 63) / 64)

  // 1) h = relu(pc @ w_pc1 + b_pc1)
  gemm_wmma_kernel<<<GRID64(N, E_DIM), blk, 0, stream>>>(
      pc, 16, w_pc1, E_DIM, b_pc1, (void*)ws_h, E_DIM, N, 16, E_DIM, 1.f, 1, 0);
  // 2) xm[:, :64] = relu(h @ w_pc2 + b_pc2)
  gemm_wmma_kernel<<<GRID64(N, 64), blk, 0, stream>>>(
      ws_h, E_DIM, w_pc2, 64, b_pc2, (void*)xm, 128, N, E_DIM, 64, 1.f, 1, 0);
  // 3) xm[:, 64:] = img_feats
  concat_img_kernel<<<dim3((N * 64 + 255) / 256), dim3(256), 0, stream>>>(img, xm, N);
  // 4) qkv = xm @ w_qkv + b_qkv
  gemm_wmma_kernel<<<GRID64(N, QKV3), blk, 0, stream>>>(
      xm, 128, w_qkv, QKV3, b_qkv, (void*)ws_qkv, QKV3, N, 128, QKV3, 1.f, 0, 0);
  // 5-7) q/k/v projections, q scaled by 1/8, f16 out
  for (int s = 0; s < 3; ++s) {
    float alpha = (s == 0) ? 0.125f : 1.f;
    gemm_wmma_kernel<<<GRID64(N, E_DIM), blk, 0, stream>>>(
        ws_qkv + s * E_DIM, QKV3, w_in + s * E_DIM, QKV3, b_in + s * E_DIM,
        (void*)(ws_qkvp + s * E_DIM), QKV3, N, E_DIM, E_DIM, alpha, 0, 1);
  }
  // 8) attention (max len 1024 -> 16 query blocks of 64)
  flash_attn_kernel<<<dim3(16, HEADS, Bn), blk, 0, stream>>>(ws_qkvp, masks, ws_ctx);
  // 9) attn_out = ctx @ w_out + b_out
  gemm_wmma_kernel<<<GRID64(N, E_DIM), blk, 0, stream>>>(
      ws_ctx, E_DIM, w_out, E_DIM, b_out, (void*)ws_ao, E_DIM, N, E_DIM, E_DIM, 1.f, 0, 0);
  // 10) preds = attn_out @ w_cls + b_cls
  gemm_wmma_kernel<<<GRID64(N, 19), blk, 0, stream>>>(
      ws_ao, E_DIM, w_cls, 19, b_cls, (void*)preds, 19, N, E_DIM, 19, 1.f, 0, 0);
#undef GRID64
}
